// Parser_91164975825447
// MI455X (gfx1250) — matrix operand order repaired and checked by reading, not compiled, not run on hardware
//
#include <hip/hip_runtime.h>
#include <hip/hip_bf16.h>
#include <stdint.h>

// ---------------------------------------------------------------------------
// Types for CDNA5 WMMA (wave32): 16x16x32 bf16 -> f32 accumulate
// ---------------------------------------------------------------------------
typedef __attribute__((ext_vector_type(16))) __bf16       v16bf;
typedef __attribute__((ext_vector_type(8)))  float        v8f;
typedef __attribute__((ext_vector_type(8)))  unsigned int v8u;

#define DD   1024
#define DD4  256
#define BB   4
#define LL   256
#define NLAYER 3
#define KW   3
#define MAXLEN 500
#define RR   8

__device__ __forceinline__ unsigned short f2bf(float f) {
    unsigned int u = __float_as_uint(f);
    unsigned int r = (u + 0x7FFFu + ((u >> 16) & 1u)) >> 16;   // RNE
    return (unsigned short)r;
}

// ---------------------------------------------------------------------------
// Elementwise converters / builders
// ---------------------------------------------------------------------------
__global__ void k_zero_us(unsigned short* __restrict__ d, int n) {
    int i = blockIdx.x * blockDim.x + threadIdx.x;
    if (i < n) d[i] = 0;
}

__global__ void k_cvt_bf16(const float* __restrict__ s, unsigned short* __restrict__ d, int n) {
    int i = blockIdx.x * blockDim.x + threadIdx.x;
    if (i < n) d[i] = f2bf(s[i]);
}

// activation (B*L x D) -> bf16, optionally zeroing rows where mask==0
__global__ void k_cvt_act(const float* __restrict__ s, const int* __restrict__ mask,
                          unsigned short* __restrict__ d, int useMask) {
    int i = blockIdx.x * blockDim.x + threadIdx.x;   // over B*L*D
    int row = i >> 10;
    float v = s[i];
    if (useMask && !mask[row]) v = 0.0f;
    d[i] = f2bf(v);
}

// conv_w (NL,D,D,K) -> per-tap N x K row-major bf16: dst[k][o*D + i] = w[layer][o][i][k]
__global__ void k_cvt_convw(const float* __restrict__ w, unsigned short* __restrict__ dst, int layer) {
    int idx = blockIdx.x * blockDim.x + threadIdx.x;  // over D*D*K
    if (idx >= DD * DD * KW) return;
    int k = idx % KW;
    int i = (idx / KW) % DD;
    int o = idx / (KW * DD);
    dst[(size_t)k * DD * DD + (size_t)o * DD + i] =
        f2bf(w[(((size_t)layer * DD + o) * DD + i) * KW + k]);
}

// Build PE table (1008 x 1024 bf16; rows >= 1001 are zero padding)
__global__ void k_build_pe(unsigned short* __restrict__ pe) {
    int idx = blockIdx.x * blockDim.x + threadIdx.x;  // over 1008*1024
    if (idx >= 1008 * DD) return;
    int p = idx / DD, d = idx % DD;
    float v = 0.0f;
    if (p < 2 * MAXLEN + 1) {
        float position = (float)(MAXLEN - p);
        float dv = __expf((float)(d & ~1) * (-9.2103403719761836f / (float)DD));
        float val = position * dv;
        v = (d & 1) ? __cosf(val) : __sinf(val);
    }
    pe[idx] = f2bf(v);
}

__global__ void k_copy(const float* __restrict__ s, float* __restrict__ d, int n) {
    int i = blockIdx.x * blockDim.x + threadIdx.x;
    if (i < n) d[i] = s[i];
}

// ---------------------------------------------------------------------------
// WMMA GEMM, 16x64 output per wave (1x4 tile blocking, A fragment reused by
// 4 WMMAs per K-step).  C(MxN,f32) = [C or bias] + A(MxK,bf16,row-shifted)*B^T.
// Bw is N x K row-major bf16 (weights are given as (N_out, K_in)).
// rowShift/rowBlock implement the conv1d tap shift with zero padding per batch;
// invalid lanes read a zeroed scratch row -> uniform control flow in the loop.
// ---------------------------------------------------------------------------
__global__ __launch_bounds__(256) void k_gemm_wmma(
    const unsigned short* __restrict__ A,
    const unsigned short* __restrict__ Bw,
    const unsigned short* __restrict__ zrow,   // Kdim zeros
    const float* __restrict__ bias,            // length N (read when !accumulate)
    float* __restrict__ C,
    int M, int N, int Kdim, int Mreal, int rowShift, int rowBlock, int accumulate)
{
    const int wave = threadIdx.x >> 5;
    const int lane = threadIdx.x & 31;
    const int ntn4 = N >> 6;                   // 64-wide N chunks
    const int tile = blockIdx.x * 8 + wave;
    const int total = (M >> 4) * ntn4;
    if (tile >= total) return;
    const int mt  = tile / ntn4;
    const int nt4 = tile % ntn4;

    const int half = lane >> 4;     // 0 or 1
    const int l15  = lane & 15;

    // A fragment (16-bit A 16x32, contiguous k order chosen by this code, same for B):
    //   lanes 0-15: M=lane, K = {0..15}; lanes 16-31: M=lane-16, K = {16..31}
    const int arow = mt * 16 + l15;
    const int srow = arow + rowShift;
    const bool avalid = (arow < Mreal) && (srow >= 0) && (srow < Mreal) &&
                        ((srow / rowBlock) == (arow / rowBlock));
    const unsigned short* aptr =
        (avalid ? (A + (size_t)srow * Kdim) : zrow) + half * 16;

    // B fragments: lane holds column N=l15, K = {0..15} / {16..31} by half
    const int nb = nt4 * 64 + l15;
    const unsigned short* bp0 = Bw + (size_t)(nb)      * Kdim + half * 16;
    const unsigned short* bp1 = Bw + (size_t)(nb + 16) * Kdim + half * 16;
    const unsigned short* bp2 = Bw + (size_t)(nb + 32) * Kdim + half * 16;
    const unsigned short* bp3 = Bw + (size_t)(nb + 48) * Kdim + half * 16;

    v8f acc0 = {}, acc1 = {}, acc2 = {}, acc3 = {};
    for (int k0 = 0; k0 < Kdim; k0 += 32) {
        const uint4 a0 = *reinterpret_cast<const uint4*>(aptr + k0);
        const uint4 a1 = *reinterpret_cast<const uint4*>(aptr + k0 + 8);
        v16bf av = __builtin_bit_cast(v16bf,
            (v8u){a0.x, a0.y, a0.z, a0.w, a1.x, a1.y, a1.z, a1.w});

        const uint4 b00 = *reinterpret_cast<const uint4*>(bp0 + k0);
        const uint4 b01 = *reinterpret_cast<const uint4*>(bp0 + k0 + 8);
        const uint4 b10 = *reinterpret_cast<const uint4*>(bp1 + k0);
        const uint4 b11 = *reinterpret_cast<const uint4*>(bp1 + k0 + 8);
        const uint4 b20 = *reinterpret_cast<const uint4*>(bp2 + k0);
        const uint4 b21 = *reinterpret_cast<const uint4*>(bp2 + k0 + 8);
        const uint4 b30 = *reinterpret_cast<const uint4*>(bp3 + k0);
        const uint4 b31 = *reinterpret_cast<const uint4*>(bp3 + k0 + 8);

        v16bf bv0 = __builtin_bit_cast(v16bf,
            (v8u){b00.x, b00.y, b00.z, b00.w, b01.x, b01.y, b01.z, b01.w});
        v16bf bv1 = __builtin_bit_cast(v16bf,
            (v8u){b10.x, b10.y, b10.z, b10.w, b11.x, b11.y, b11.z, b11.w});
        v16bf bv2 = __builtin_bit_cast(v16bf,
            (v8u){b20.x, b20.y, b20.z, b20.w, b21.x, b21.y, b21.z, b21.w});
        v16bf bv3 = __builtin_bit_cast(v16bf,
            (v8u){b30.x, b30.y, b30.z, b30.w, b31.x, b31.y, b31.z, b31.w});

        acc0 = __builtin_amdgcn_wmma_f32_16x16x32_bf16(false, av, false, bv0, (short)0, acc0, false, false);
        acc1 = __builtin_amdgcn_wmma_f32_16x16x32_bf16(false, av, false, bv1, (short)0, acc1, false, false);
        acc2 = __builtin_amdgcn_wmma_f32_16x16x32_bf16(false, av, false, bv2, (short)0, acc2, false, false);
        acc3 = __builtin_amdgcn_wmma_f32_16x16x32_bf16(false, av, false, bv3, (short)0, acc3, false, false);
    }

    // C/D layout: VGPR r: lanes 0-15 -> (M=r, N=lane); lanes 16-31 -> (M=8+r, N=lane-16)
    const int rbase = mt * 16 + half * 8;
    #pragma unroll
    for (int t = 0; t < 4; ++t) {
        const v8f acc = (t == 0) ? acc0 : (t == 1) ? acc1 : (t == 2) ? acc2 : acc3;
        const int col = nt4 * 64 + t * 16 + l15;
        #pragma unroll
        for (int r = 0; r < 8; ++r) {
            const int row = rbase + r;
            if (row < Mreal) {
                const size_t o = (size_t)row * N + col;
                C[o] = acc[r] + (accumulate ? C[o] : bias[col]);
            }
        }
    }
}

// ---------------------------------------------------------------------------
// LayerNorm (no affine) + tanh over rows of 1024 (conv stack activation)
// ---------------------------------------------------------------------------
__global__ __launch_bounds__(256) void k_ln_tanh_1024(const float* __restrict__ y,
                                                      float* __restrict__ h) {
    __shared__ float r1[256];
    __shared__ float r2[256];
    const int row = blockIdx.x;
    const int t = threadIdx.x;
    const float* yr = y + (size_t)row * DD;
    float4 v = *reinterpret_cast<const float4*>(yr + t * 4);
    r1[t] = v.x + v.y + v.z + v.w;
    r2[t] = v.x * v.x + v.y * v.y + v.z * v.z + v.w * v.w;
    __syncthreads();
    for (int o = 128; o >= 1; o >>= 1) {
        if (t < o) { r1[t] += r1[t + o]; r2[t] += r2[t + o]; }
        __syncthreads();
    }
    const float mu = r1[0] * (1.0f / DD);
    const float var = r2[0] * (1.0f / DD) - mu * mu;
    const float rs = rsqrtf(var + 1e-5f);
    float4 o4;
    o4.x = tanhf((v.x - mu) * rs);
    o4.y = tanhf((v.y - mu) * rs);
    o4.z = tanhf((v.z - mu) * rs);
    o4.w = tanhf((v.w - mu) * rs);
    *reinterpret_cast<float4*>(h + (size_t)row * DD + t * 4) = o4;
}

// ---------------------------------------------------------------------------
// cum -> f_cumul / b_cumul  (reference zeroes cum where mask is TRUE)
// ---------------------------------------------------------------------------
__global__ void k_cumsum(const float* __restrict__ cum, const int* __restrict__ mask,
                         float* __restrict__ fc, float* __restrict__ bc) {
    int tid = blockIdx.x * blockDim.x + threadIdx.x;   // B*D4
    if (tid >= BB * DD4) return;
    int b = tid / DD4, k = tid % DD4;
    float run = 0.0f;
    for (int l = 0; l < LL; ++l) {
        int row = b * LL + l;
        float c = mask[row] ? 0.0f : cum[(size_t)row * DD4 + k];
        fc[(size_t)row * DD4 + k] = run;
        run += c;
    }
    const float total = run;
    run = 0.0f;
    for (int l = 0; l < LL; ++l) {
        int row = b * LL + l;
        float c = mask[row] ? 0.0f : cum[(size_t)row * DD4 + k];
        run += c;
        bc[(size_t)row * DD4 + k] = total - run;
    }
}

// ---------------------------------------------------------------------------
// Fused pairwise stage: one wave32 per (b,i,j) row of length D4=256.
// hid = c_lin[b,i] + p_lin[b,j] + f_cumul[b,i] - b_cumul[b,j] + pos[500+i-j]
// -> LN -> *g+b -> tanh -> dot Wo (+bo) -> 7.5*tanh(x/7.5) -> mask -> logits/root
// ---------------------------------------------------------------------------
__global__ __launch_bounds__(256) void k_hid(
    const float* __restrict__ c_lin, const float* __restrict__ p_lin,
    const float* __restrict__ fcum,  const float* __restrict__ bcum,
    const float* __restrict__ pos,   const float* __restrict__ ln_g,
    const float* __restrict__ ln_b,  const float* __restrict__ Wo,
    const float* __restrict__ bo,    const int* __restrict__ mask,
    float* __restrict__ logits, float* __restrict__ root)
{
    __shared__ float sWo[RR * DD4];
    __shared__ float sg[DD4];
    __shared__ float sb[DD4];
    for (int i = threadIdx.x; i < RR * DD4; i += 256) sWo[i] = Wo[i];
    sg[threadIdx.x] = ln_g[threadIdx.x];
    sb[threadIdx.x] = ln_b[threadIdx.x];
    __syncthreads();

    const int wave = threadIdx.x >> 5;
    const int lane = threadIdx.x & 31;
    const long rowId = (long)blockIdx.x * 8 + wave;          // < B*L*L
    const int j = (int)(rowId & 255);
    const int i = (int)((rowId >> 8) & 255);
    const int b = (int)(rowId >> 16);

    const float* cr = c_lin + ((size_t)(b * LL + i)) * DD4;
    const float* pr = p_lin + ((size_t)(b * LL + j)) * DD4;
    const float* fr = fcum  + ((size_t)(b * LL + i)) * DD4;
    const float* br = bcum  + ((size_t)(b * LL + j)) * DD4;
    const float* qr = pos   + ((size_t)(MAXLEN + i - j)) * DD4;

    const int d0 = lane * 8;
    float s[8];
    {
        float4 a0 = *reinterpret_cast<const float4*>(cr + d0);
        float4 a1 = *reinterpret_cast<const float4*>(cr + d0 + 4);
        float4 b0 = *reinterpret_cast<const float4*>(pr + d0);
        float4 b1 = *reinterpret_cast<const float4*>(pr + d0 + 4);
        float4 c0 = *reinterpret_cast<const float4*>(fr + d0);
        float4 c1 = *reinterpret_cast<const float4*>(fr + d0 + 4);
        float4 e0 = *reinterpret_cast<const float4*>(br + d0);
        float4 e1 = *reinterpret_cast<const float4*>(br + d0 + 4);
        float4 q0 = *reinterpret_cast<const float4*>(qr + d0);
        float4 q1 = *reinterpret_cast<const float4*>(qr + d0 + 4);
        s[0] = a0.x + b0.x + c0.x - e0.x + q0.x;
        s[1] = a0.y + b0.y + c0.y - e0.y + q0.y;
        s[2] = a0.z + b0.z + c0.z - e0.z + q0.z;
        s[3] = a0.w + b0.w + c0.w - e0.w + q0.w;
        s[4] = a1.x + b1.x + c1.x - e1.x + q1.x;
        s[5] = a1.y + b1.y + c1.y - e1.y + q1.y;
        s[6] = a1.z + b1.z + c1.z - e1.z + q1.z;
        s[7] = a1.w + b1.w + c1.w - e1.w + q1.w;
    }

    float sum = 0.0f, sq = 0.0f;
    #pragma unroll
    for (int t = 0; t < 8; ++t) { sum += s[t]; sq += s[t] * s[t]; }
    #pragma unroll
    for (int o = 16; o >= 1; o >>= 1) {
        sum += __shfl_xor(sum, o, 32);
        sq  += __shfl_xor(sq,  o, 32);
    }
    const float mu  = sum * (1.0f / DD4);
    const float var = sq * (1.0f / DD4) - mu * mu;
    const float rs  = rsqrtf(var + 1e-5f);

    float part[RR];
    #pragma unroll
    for (int r = 0; r < RR; ++r) part[r] = 0.0f;
    #pragma unroll
    for (int t = 0; t < 8; ++t) {
        const int d = d0 + t;
        const float tv = tanhf((s[t] - mu) * rs * sg[d] + sb[d]);
        #pragma unroll
        for (int r = 0; r < RR; ++r) part[r] += tv * sWo[r * DD4 + d];
    }
    #pragma unroll
    for (int r = 0; r < RR; ++r) {
        #pragma unroll
        for (int o = 16; o >= 1; o >>= 1) part[r] += __shfl_xor(part[r], o, 32);
    }

    const bool vis = mask[b * LL + i] && mask[b * LL + j];
    if (lane < RR) {
        float v = tanhf((part[lane] + bo[lane]) * (1.0f / 7.5f)) * 7.5f;
        if (!vis) v = -64.0f;
        logits[rowId * RR + lane] = v;
        if (i == j) root[((long)(b * LL + i)) * RR + lane] = v;
    }
}

// ---------------------------------------------------------------------------
// Host-side orchestration
// ---------------------------------------------------------------------------
static inline int nblk(long n, int b) { return (int)((n + b - 1) / b); }

extern "C" void kernel_launch(void* const* d_in, const int* in_sizes, int n_in,
                              void* d_out, int out_size, void* d_ws, size_t ws_size,
                              hipStream_t stream) {
    const float* X      = (const float*)d_in[0];
    const float* conv_w = (const float*)d_in[1];
    const float* conv_b = (const float*)d_in[2];
    const float* Wc     = (const float*)d_in[3];
    const float* bc     = (const float*)d_in[4];
    const float* Wp     = (const float*)d_in[5];
    const float* bp     = (const float*)d_in[6];
    const float* Wcum   = (const float*)d_in[7];
    const float* bcum   = (const float*)d_in[8];
    const float* Wr     = (const float*)d_in[9];
    const float* br     = (const float*)d_in[10];
    const float* ln_g   = (const float*)d_in[11];
    const float* ln_b   = (const float*)d_in[12];
    const float* Wo     = (const float*)d_in[13];
    const float* bo     = (const float*)d_in[14];
    const int*   mask   = (const int*)d_in[15];

    float* out_logits = (float*)d_out;                      // B*L*L*R
    float* out_root   = out_logits + (size_t)BB * LL * LL * RR;
    float* out_h      = out_root + (size_t)BB * LL * RR;

    char* ws = (char*)d_ws;
    const int ML = BB * LL;  // 1024 rows
    size_t off = 0;
    auto alloc = [&](size_t bytes) { size_t o = off; off += (bytes + 255) & ~(size_t)255; return o; };

    unsigned short* Xbf   = (unsigned short*)(ws + alloc((size_t)ML * DD * 2));
    unsigned short* hbf   = (unsigned short*)(ws + alloc((size_t)ML * DD * 2));
    unsigned short* cwbf  = (unsigned short*)(ws + alloc((size_t)KW * DD * DD * 2));
    unsigned short* Wcbf  = (unsigned short*)(ws + alloc((size_t)DD4 * DD * 2));
    unsigned short* Wpbf  = (unsigned short*)(ws + alloc((size_t)DD4 * DD * 2));
    unsigned short* Wcmbf = (unsigned short*)(ws + alloc((size_t)DD4 * DD * 2));
    unsigned short* Wrbf  = (unsigned short*)(ws + alloc((size_t)DD4 * DD * 2));
    unsigned short* PEbf  = (unsigned short*)(ws + alloc((size_t)1008 * DD * 2));
    unsigned short* zrow  = (unsigned short*)(ws + alloc((size_t)DD * 2));
    float* hbuf  = (float*)(ws + alloc((size_t)ML * DD * 4));
    float* ybuf  = (float*)(ws + alloc((size_t)ML * DD * 4));
    float* cum   = (float*)(ws + alloc((size_t)ML * DD4 * 4));
    float* clin  = (float*)(ws + alloc((size_t)ML * DD4 * 4));
    float* plin  = (float*)(ws + alloc((size_t)ML * DD4 * 4));
    float* fcum  = (float*)(ws + alloc((size_t)ML * DD4 * 4));
    float* bcum_ = (float*)(ws + alloc((size_t)ML * DD4 * 4));
    float* posb  = (float*)(ws + alloc((size_t)1008 * DD4 * 4));
    (void)ws_size; (void)n_in; (void)in_sizes; (void)out_size;

    const int BIG = 1 << 30;

    // --- one-time (per call) conversions ---
    k_zero_us<<<nblk(DD, 256), 256, 0, stream>>>(zrow, DD);
    k_cvt_bf16<<<nblk((long)ML * DD, 256), 256, 0, stream>>>(X, Xbf, ML * DD);
    k_cvt_bf16<<<nblk((long)DD4 * DD, 256), 256, 0, stream>>>(Wc,   Wcbf,  DD4 * DD);
    k_cvt_bf16<<<nblk((long)DD4 * DD, 256), 256, 0, stream>>>(Wp,   Wpbf,  DD4 * DD);
    k_cvt_bf16<<<nblk((long)DD4 * DD, 256), 256, 0, stream>>>(Wcum, Wcmbf, DD4 * DD);
    k_cvt_bf16<<<nblk((long)DD4 * DD, 256), 256, 0, stream>>>(Wr,   Wrbf,  DD4 * DD);
    k_build_pe<<<nblk((long)1008 * DD, 256), 256, 0, stream>>>(PEbf);

    // --- cum = X @ Wcum^T + bcum ; then masked prefix/suffix sums ---
    {
        int tiles = (ML / 16) * (DD4 / 64);
        k_gemm_wmma<<<nblk(tiles, 8), 256, 0, stream>>>(Xbf, Wcmbf, zrow, bcum, cum,
                                                        ML, DD4, DD, ML, 0, BIG, 0);
    }
    k_cumsum<<<nblk(BB * DD4, 256), 256, 0, stream>>>(cum, mask, fcum, bcum_);

    // --- conv stack: 3 layers x 3 shifted GEMM taps + LN/tanh ---
    for (int layer = 0; layer < NLAYER; ++layer) {
        k_cvt_convw<<<nblk((long)DD * DD * KW, 256), 256, 0, stream>>>(conv_w, cwbf, layer);
        const float* src = (layer == 0) ? X : hbuf;
        k_cvt_act<<<nblk((long)ML * DD, 256), 256, 0, stream>>>(src, mask, hbf, 1);
        int tiles = (ML / 16) * (DD / 64);
        for (int k = 0; k < KW; ++k) {
            k_gemm_wmma<<<nblk(tiles, 8), 256, 0, stream>>>(
                hbf, cwbf + (size_t)k * DD * DD, zrow, conv_b + (size_t)layer * DD, ybuf,
                ML, DD, DD, ML, k - 1, LL, (k > 0) ? 1 : 0);
        }
        k_ln_tanh_1024<<<ML, 256, 0, stream>>>(ybuf, hbuf);
    }

    // --- final h -> bf16 (unmasked), projections, positional table ---
    k_cvt_act<<<nblk((long)ML * DD, 256), 256, 0, stream>>>(hbuf, mask, hbf, 0);
    {
        int tiles = (ML / 16) * (DD4 / 64);
        k_gemm_wmma<<<nblk(tiles, 8), 256, 0, stream>>>(hbf, Wcbf, zrow, bc, clin,
                                                        ML, DD4, DD, ML, 0, BIG, 0);
        k_gemm_wmma<<<nblk(tiles, 8), 256, 0, stream>>>(hbf, Wpbf, zrow, bp, plin,
                                                        ML, DD4, DD, ML, 0, BIG, 0);
    }
    {
        int tiles = (1008 / 16) * (DD4 / 64);
        k_gemm_wmma<<<nblk(tiles, 8), 256, 0, stream>>>(PEbf, Wrbf, zrow, br, posb,
                                                        1008, DD4, DD, 2 * MAXLEN + 1, 0, BIG, 0);
    }

    // --- h output ---
    k_copy<<<nblk((long)ML * DD, 256), 256, 0, stream>>>(hbuf, out_h, ML * DD);

    // --- fused pairwise stage: 262144 rows, 1 wave each, 8 waves/block ---
    k_hid<<<(BB * LL * LL) / 8, 256, 0, stream>>>(clin, plin, fcum, bcum_, posb,
                                                  ln_g, ln_b, Wo, bo, mask,
                                                  out_logits, out_root);
}
